// ExplainModule_89739046683412
// MI455X (gfx1250) — matrix operand order repaired and checked by reading, not compiled, not run on hardware
//
#include <hip/hip_runtime.h>

typedef __attribute__((ext_vector_type(16))) _Float16 v16h;
typedef __attribute__((ext_vector_type(8)))  float    v8f;

#define NN 1024
#define DD 64
#define HH 64

// ---------------------------------------------------------------------------
// Stage 1: pre = embed @ W1[:D] + b1 ; post = embed @ W1[D:]
// One wave (32 threads) computes one 16x16 f32 tile with two
// v_wmma_f32_16x16x32_f16 ops (K=64 split into 2 chunks of 32).
// Register layouts per CDNA5 ISA sec 7.12.2 (wave32).
// ---------------------------------------------------------------------------
__global__ __launch_bounds__(32) void wmma_pre_post(
    const float* __restrict__ embed, const float* __restrict__ W1,
    const float* __restrict__ b1, float* __restrict__ pre,
    float* __restrict__ post)
{
    const int lane  = threadIdx.x;        // 0..31
    const int tm    = blockIdx.x;         // M tile: 0..63
    const int tn    = blockIdx.y;         // N tile: 0..3
    const int sel   = blockIdx.z;         // 0 -> pre (W1 rows 0..63), 1 -> post
    const int halfL = lane >> 4;          // lane half
    const int l16   = lane & 15;

    const int rowA = tm * 16 + l16;       // M index for A / D
    const int colB = tn * 16 + l16;       // N index for B / D
    const float* wbase = W1 + sel * (DD * HH);

    v8f c = {};
#pragma unroll
    for (int kc = 0; kc < 2; ++kc) {
        // --- A tile 16x32 f16: elems 0..7 hold K = 8*halfL + e,
        //     elems 8..15 hold K = 16 + 8*halfL + (e-8)  (all + kc*32)
        v16h a;
        {
            const float* arow = embed + rowA * DD + kc * 32 + 8 * halfL;
            float4 g0 = *(const float4*)(arow + 0);
            float4 g1 = *(const float4*)(arow + 4);
            float4 g2 = *(const float4*)(arow + 16);
            float4 g3 = *(const float4*)(arow + 20);
            a[0]=(_Float16)g0.x;  a[1]=(_Float16)g0.y;  a[2]=(_Float16)g0.z;  a[3]=(_Float16)g0.w;
            a[4]=(_Float16)g1.x;  a[5]=(_Float16)g1.y;  a[6]=(_Float16)g1.z;  a[7]=(_Float16)g1.w;
            a[8]=(_Float16)g2.x;  a[9]=(_Float16)g2.y;  a[10]=(_Float16)g2.z; a[11]=(_Float16)g2.w;
            a[12]=(_Float16)g3.x; a[13]=(_Float16)g3.y; a[14]=(_Float16)g3.z; a[15]=(_Float16)g3.w;
        }
        // --- B tile 32x16 f16: elems 0..7 hold K = 8*halfL + e, elems 8..15 hold K = 16 + 8*halfL + (e-8) (+kc*32), N = lane%16
        v16h b;
        {
            const float* bcol = wbase + (kc * 32 + 8 * halfL) * HH + colB;
#pragma unroll
            for (int e = 0; e < 16; ++e) b[e] = (_Float16)bcol[((e & 7) + (e >> 3) * 16) * HH];
        }
        c = __builtin_amdgcn_wmma_f32_16x16x32_f16(
                /*neg_a=*/false, a, /*neg_b=*/false, b,
                /*c_mod=*/(short)0, c, /*reuse_a=*/false, /*reuse_b=*/false);
    }

    // D layout: VGPR r -> M = r + 8*halfL, N = lane%16. Fold b1 into pre.
    float* dst  = sel ? post : pre;
    float  bias = sel ? 0.0f : b1[colB];
#pragma unroll
    for (int r = 0; r < 8; ++r) {
        int m = r + 8 * halfL;
        dst[(tm * 16 + m) * HH + colB] = c[r] + bias;
    }
}

// ---------------------------------------------------------------------------
// Stage 2: gate[i,j] = sigmoid( log(u/(1-u)) + b2 +
//                               sum_h relu(preB[i,h] + post[j,h]) * w2[h] )
// 256-thread block computes a 32x32 (i,j) tile; each thread does 4 pairs
// sharing one pre-row kept in VGPRs; post rows streamed from padded LDS.
// ---------------------------------------------------------------------------
#define TI 32
#define TJ 32
#define LSTR 68   // padded LDS row stride (floats); keeps b128 reads conflict-lean

__global__ __launch_bounds__(256) void gate_kernel(
    const float* __restrict__ preB, const float* __restrict__ post,
    const float* __restrict__ u, const float* __restrict__ W2,
    const float* __restrict__ b2, float* __restrict__ gate)
{
    __shared__ float sp[TI * LSTR];
    __shared__ float sq[TJ * LSTR];
    __shared__ float sw[HH];

    const int t  = threadIdx.x;
    const int i0 = blockIdx.y * TI;
    const int j0 = blockIdx.x * TJ;

    // Cooperative tile load: 512 float4 chunks each for pre/post.
#pragma unroll
    for (int c = t; c < TI * 16; c += 256) {
        int r = c >> 4, q = c & 15;
        float4 vp = *(const float4*)(preB + (i0 + r) * HH + q * 4);
        float4 vq = *(const float4*)(post + (j0 + r) * HH + q * 4);
        *(float4*)(sp + r * LSTR + q * 4) = vp;
        *(float4*)(sq + r * LSTR + q * 4) = vq;
    }
    if (t < 16) *(float4*)(sw + t * 4) = *(const float4*)(W2 + t * 4);

    const int ti = t >> 3;   // 0..31: i within tile
    const int jg = t & 7;    // 0..7 : j group; thread's j = j0 + jg + 8*jj

    // Warm L2/L0 for the epilogue's u reads while we compute.
    __builtin_prefetch(u + (i0 + ti) * NN + j0, 0, 1);

    __syncthreads();

    // Pre row -> registers (64 floats).
    float4 P[16];
#pragma unroll
    for (int q = 0; q < 16; ++q) P[q] = *(const float4*)(sp + ti * LSTR + q * 4);

    float acc[4] = {0.f, 0.f, 0.f, 0.f};
#pragma unroll
    for (int q = 0; q < 16; ++q) {
        float4 w = *(const float4*)(sw + q * 4);   // uniform -> LDS broadcast
        float4 p = P[q];
#pragma unroll
        for (int jj = 0; jj < 4; ++jj) {
            float4 v = *(const float4*)(sq + (jg + 8 * jj) * LSTR + q * 4);
            float s0 = fmaxf(p.x + v.x, 0.f);
            float s1 = fmaxf(p.y + v.y, 0.f);
            float s2 = fmaxf(p.z + v.z, 0.f);
            float s3 = fmaxf(p.w + v.w, 0.f);
            acc[jj] = fmaf(s0, w.x, acc[jj]);
            acc[jj] = fmaf(s1, w.y, acc[jj]);
            acc[jj] = fmaf(s2, w.z, acc[jj]);
            acc[jj] = fmaf(s3, w.w, acc[jj]);
        }
    }

    const float bb = b2[0];
    const int   i  = i0 + ti;
#pragma unroll
    for (int jj = 0; jj < 4; ++jj) {
        int   j     = j0 + jg + 8 * jj;
        float la    = acc[jj] + bb;
        float uu    = u[i * NN + j];
        float noise = __logf(uu) - log1pf(-uu);
        float s     = noise + la;                     // beta = 1
        float g     = 1.0f / (1.0f + __expf(-s));     // stable both tails
        gate[i * NN + j] = g;
    }
}

// ---------------------------------------------------------------------------
// Stage 3: out = adj * 0.5*(gate + gate^T) via LDS transpose tile.
// ---------------------------------------------------------------------------
__global__ __launch_bounds__(256) void sym_mask_kernel(
    const float* __restrict__ gate, const float* __restrict__ adj,
    float* __restrict__ out)
{
    __shared__ float tl[16][17];
    const int tx = threadIdx.x, ty = threadIdx.y;
    const int i0 = blockIdx.y * 16, j0 = blockIdx.x * 16;

    tl[ty][tx] = gate[(j0 + ty) * NN + (i0 + tx)];
    __syncthreads();

    const int i = i0 + ty, j = j0 + tx;
    float s = 0.5f * (gate[i * NN + j] + tl[tx][ty]);
    out[i * NN + j] = adj[i * NN + j] * s;
}

// ---------------------------------------------------------------------------
extern "C" void kernel_launch(void* const* d_in, const int* in_sizes, int n_in,
                              void* d_out, int out_size, void* d_ws, size_t ws_size,
                              hipStream_t stream)
{
    (void)in_sizes; (void)n_in; (void)out_size; (void)ws_size;
    const float* embed = (const float*)d_in[0];  // [1024,64]
    const float* adj   = (const float*)d_in[1];  // [1024,1024]
    const float* u     = (const float*)d_in[2];  // [1024*1024]
    const float* W1    = (const float*)d_in[3];  // [128,64]
    const float* b1    = (const float*)d_in[4];  // [64]
    const float* W2    = (const float*)d_in[5];  // [64,1]
    const float* b2    = (const float*)d_in[6];  // [1]
    float* out = (float*)d_out;

    float* pre  = (float*)d_ws;                  // [1024,64]  (includes +b1)
    float* post = pre  + NN * HH;                // [1024,64]
    float* gate = post + NN * HH;                // [1024,1024]

    // Stage 1: WMMA GEMMs (pre & post), one wave per 16x16 tile.
    wmma_pre_post<<<dim3(NN / 16, HH / 16, 2), 32, 0, stream>>>(
        embed, W1, b1, pre, post);

    // Stage 2: fused relu-dot + concrete gate.
    gate_kernel<<<dim3(NN / TJ, NN / TI), 256, 0, stream>>>(
        pre, post, u, W2, b2, gate);

    // Stage 3: symmetrize + mask.
    sym_mask_kernel<<<dim3(NN / 16, NN / 16), dim3(16, 16), 0, stream>>>(
        gate, adj, out);
}
